// Body3_interaction_MACE_15367392985804
// MI455X (gfx1250) — compile-verified, matrix operand order checked by reading, not run on hardware
//
#include <hip/hip_runtime.h>
#include <hip/hip_bf16.h>

// ---------------------------------------------------------------------------
// MACE body-3 interaction for MI455X (gfx1250), wave32 + WMMA bf16.
// Dominant cost reformulated as GEMM: D[nc,oab] = Z[nc,ip] * U3ext[ip,oab]
//   M=65536, Ncols=729(->736), K=148(->160), with c2 folded into K and the
//   two trailing v-contractions fused as an LDS epilogue.
// ---------------------------------------------------------------------------

typedef __attribute__((ext_vector_type(16))) __bf16 v16bf;
typedef __attribute__((ext_vector_type(8)))  float  v8f;
typedef __attribute__((ext_vector_type(4)))  float  v4f;
typedef __attribute__((ext_vector_type(8)))  short  short8;
typedef __attribute__((ext_vector_type(16))) short  short16;

union BfPack { short16 s; v16bf b; };

__device__ __forceinline__ short f2bf(float f) {
    union { float f; unsigned u; } x; x.f = f;
    unsigned r = x.u + 0x7FFFu + ((x.u >> 16) & 1u);
    return (short)(r >> 16);
}

#define NN     1024
#define CC     64
#define MM     9          // (LMAX+1)^2
#define NCOLS  729        // 9^3
#define NCOLP  736        // padded to 46*16
#define NK     160        // 144 (U3) + 4 (U2) padded to 5*32
#define NKREAL 148
#define ASTR   168        // LDS A row stride (shorts), 16B aligned, bank-skewed

// ---------------------------------------------------------------------------
// Kernel 0a: build B matrix (bf16) in workspace: B[col][k], col=(o*81+a*9+b)
//   k<144 : U3[o,a,b,i,p] with k=i*16+p     (U3 flat is already col*144+k)
//   144..147 : U2[o,a,b,p]                  (U2 flat is col*4+p)
//   else 0 (K padding / column padding)
// ---------------------------------------------------------------------------
__global__ __launch_bounds__(256)
void build_B(const float* __restrict__ U3, const float* __restrict__ U2,
             short* __restrict__ B)
{
    int gid = blockIdx.x * 256 + threadIdx.x;
    if (gid >= NCOLP * NK) return;
    int col = gid / NK, k = gid % NK;
    float v = 0.f;
    if (col < NCOLS) {
        if (k < 144)            v = U3[col * 144 + k];
        else if (k < NKREAL)    v = U2[col * 4 + (k - 144)];
    }
    B[col * NK + k] = f2bf(v);
}

// ---------------------------------------------------------------------------
// Kernel 0b: collapse trailing so3_linears: w_comb[l] = w_fc2[l] @ w_lin[l]
// ---------------------------------------------------------------------------
__global__ __launch_bounds__(256)
void build_wcomb(const float* __restrict__ w_lin, const float* __restrict__ w_fc2,
                 float* __restrict__ wcomb)
{
    int gid = blockIdx.x * 256 + threadIdx.x;
    if (gid >= 3 * 64 * 64) return;
    int l = gid / 4096, e = (gid / 64) % 64, c = gid % 64;
    const float* f2 = w_fc2 + (l * 64 + e) * 64;   // f2[d]
    const float* fl = w_lin + l * 64 * 64 + c;     // fl[d*64]
    float s = 0.f;
#pragma unroll 8
    for (int d = 0; d < 64; ++d) s += f2[d] * fl[d * 64];
    wcomb[gid] = s;
}

// ---------------------------------------------------------------------------
// Kernel 1: front so3_linear (fc1), output transposed: xw[n][c][m] (f32)
// ---------------------------------------------------------------------------
__global__ __launch_bounds__(256)
void fc1_kernel(const float* __restrict__ X, const float* __restrict__ w_fc1,
                const float* __restrict__ b_fc1, float* __restrict__ xw)
{
    const int n = blockIdx.x;
    __shared__ float xs[MM * CC];
    for (int idx = threadIdx.x; idx < MM * CC; idx += 256)
        xs[idx] = X[n * MM * CC + idx];
    __syncthreads();
    for (int idx = threadIdx.x; idx < CC * MM; idx += 256) {
        int d = idx / MM, m = idx % MM;
        int l = (m == 0) ? 0 : (m < 4 ? 1 : 2);
        const float* wrow = w_fc1 + (l * 64 + d) * 64;
        float s = (m == 0) ? b_fc1[d] : 0.f;
#pragma unroll 8
        for (int c = 0; c < 64; ++c) s += xs[m * 64 + c] * wrow[c];
        xw[(n * CC + d) * MM + m] = s;
    }
}

// ---------------------------------------------------------------------------
// Kernel 2 (main): per workgroup -> 16 rows (fixed n, c0..c0+15), all 729 cols.
// 4 waves; each wave owns column-tiles ct, ct+4, ... (46 tiles of 16 cols).
// GEMM: 5 x v_wmma_f32_16x16x32_bf16 per tile; epilogue contracts b (via a
// precomputed col-decode table + vectorized ds_load_b128 of xvT) then a.
// ---------------------------------------------------------------------------
__global__ __launch_bounds__(128)
void mace_body3_main(const int*   __restrict__ anum,
                     const float* __restrict__ W3,
                     const float* __restrict__ W2,
                     const float* __restrict__ W1,
                     const float* __restrict__ U1,
                     const float* __restrict__ xw,
                     const short* __restrict__ B,
                     float*       __restrict__ t)
{
    const int wg  = blockIdx.x;          // 0..4095
    const int n   = wg >> 2;
    const int c0  = (wg & 3) << 4;
    const int tid = threadIdx.x;
    const int an  = anum[n];

    __shared__ __align__(16) short Ash[16 * ASTR];  // A operand, bf16 bits
    __shared__ float xv[16 * MM];                   // v[r][i]   (A build)
    __shared__ __align__(16) float xvT[MM * 16];    // v[i][r]   (epilogues)
    __shared__ float w3s[16 * 16];                  // [p][r]
    __shared__ float w2s[4 * 16];                   // [p][r]
    __shared__ float w1s[2 * 16];                   // [p][r]
    __shared__ float out_oa[16 * 81];               // per-row (o,a) accums
    __shared__ unsigned tab[NCOLP];                 // col -> (oa<<16)|b2

    // ---- stage per-row operands into LDS -------------------------------
    for (int idx = tid; idx < 16 * MM; idx += 128) {
        int r = idx / MM, i = idx % MM;
        float v = xw[(n * CC + c0 + r) * MM + i];
        xv[idx] = v;
        xvT[i * 16 + r] = v;
    }
    for (int idx = tid; idx < 256; idx += 128) {
        int p = idx >> 4, r = idx & 15;
        w3s[idx] = W3[(an * 16 + p) * 64 + c0 + r];
    }
    if (tid < 64)  { int p = tid >> 4, r = tid & 15; w2s[tid] = W2[(an * 4 + p) * 64 + c0 + r]; }
    if (tid < 32)  { int p = tid >> 4, r = tid & 15; w1s[tid] = W1[(an * 2 + p) * 64 + c0 + r]; }
    for (int idx = tid; idx < 16 * 81; idx += 128) out_oa[idx] = 0.f;
    for (int idx = tid; idx < NCOLP; idx += 128) {
        unsigned e = 0xFFFFFFFFu;
        if (idx < NCOLS) {
            int o = idx / 81, a2 = (idx / 9) % 9, b2 = idx % 9;
            e = ((unsigned)(o * 9 + a2) << 16) | (unsigned)b2;
        }
        tab[idx] = e;
    }
    __syncthreads();

    // ---- build A (Z matrix) in bf16: A[r][k] = v[i]*w3[p] | w2[p] | 0 --
    for (int idx = tid; idx < 16 * NK; idx += 128) {
        int r = idx / NK, k = idx % NK;
        float v;
        if (k < 144)          v = xv[r * MM + (k >> 4)] * w3s[(k & 15) * 16 + r];
        else if (k < NKREAL)  v = w2s[(k - 144) * 16 + r];
        else                  v = 0.f;
        Ash[r * ASTR + k] = f2bf(v);
    }
    __syncthreads();

    const int lane = tid & 31;
    const int wid  = tid >> 5;
    const int l15  = lane & 15;
    const int hi   = lane >> 4;   // 0/1: which K half-group this lane carries
    const int rb   = hi << 3;     // first of the 8 C/D rows this lane owns

    for (int ct = wid; ct < 46; ct += 4) {
        const int col = ct * 16 + l15;
        const short* bptr = B + col * NK;
        if (ct + 4 < 46)
            __builtin_prefetch(B + ((ct + 4) * 16 + l15) * NK, 0, 1);

        v8f acc = (v8f){0.f, 0.f, 0.f, 0.f, 0.f, 0.f, 0.f, 0.f};
#pragma unroll
        for (int ks = 0; ks < 5; ++ks) {
            // CDNA5 16-bit A/B fragment: lanes 0-15 rows/cols, elems 0..7 are
            // K = 8*hi..8*hi+7, elems 8..15 are K = 16+8*hi..16+8*hi+7.
            const int off0 = ks * 32 + hi * 8;
            short8 alo = *(const short8*)(&Ash[l15 * ASTR + off0]);
            short8 ahi = *(const short8*)(&Ash[l15 * ASTR + off0 + 16]);
            short8 blo = *(const short8*)(bptr + off0);
            short8 bhi = *(const short8*)(bptr + off0 + 16);
            BfPack a, b;
            a.s = __builtin_shufflevector(alo, ahi, 0,1,2,3,4,5,6,7,8,9,10,11,12,13,14,15);
            b.s = __builtin_shufflevector(blo, bhi, 0,1,2,3,4,5,6,7,8,9,10,11,12,13,14,15);
            acc = __builtin_amdgcn_wmma_f32_16x16x32_bf16(
                      false, a.b, false, b.b, (short)0, acc, false, false);
        }
        // epilogue: (c2 + out5)[.,col] * v[b]  ->  out_oa[row][o*9+a]
        const unsigned e = tab[col];
        if (e != 0xFFFFFFFFu) {
            const int oa = (int)(e >> 16);
            const int b2 = (int)(e & 0xFFFFu);
            // 8 consecutive rows of v[b2]: two ds_load_b128, one wait
            const v4f xb0 = *(const v4f*)&xvT[b2 * 16 + rb];
            const v4f xb1 = *(const v4f*)&xvT[b2 * 16 + rb + 4];
            float* dst = &out_oa[rb * 81 + oa];
            atomicAdd(dst + 0 * 81, acc[0] * xb0.x);
            atomicAdd(dst + 1 * 81, acc[1] * xb0.y);
            atomicAdd(dst + 2 * 81, acc[2] * xb0.z);
            atomicAdd(dst + 3 * 81, acc[3] * xb0.w);
            atomicAdd(dst + 4 * 81, acc[4] * xb1.x);
            atomicAdd(dst + 5 * 81, acc[5] * xb1.y);
            atomicAdd(dst + 6 * 81, acc[6] * xb1.z);
            atomicAdd(dst + 7 * 81, acc[7] * xb1.w);
        }
    }
    __syncthreads();

    // ---- final: out[o] = sum_a (c1[o,a] + out_oa[o,a]) * v[a] ----------
    for (int idx = tid; idx < 16 * MM; idx += 128) {
        int r = idx / MM, o = idx % MM;
        float s = 0.f;
#pragma unroll
        for (int a = 0; a < 9; ++a) {
            float c1 = U1[(o * 9 + a) * 2] * w1s[r]
                     + U1[(o * 9 + a) * 2 + 1] * w1s[16 + r];
            s += (c1 + out_oa[r * 81 + o * 9 + a]) * xvT[a * 16 + r];
        }
        t[(n * CC + c0 + r) * MM + o] = s;
    }
}

// ---------------------------------------------------------------------------
// Kernel 3: y[n,m,e] = sum_c t[n,c,m] * wcomb[l(m)][e,c]  (+ b_fc2, m==0)
// ---------------------------------------------------------------------------
__global__ __launch_bounds__(256)
void tail_kernel(const float* __restrict__ t, const float* __restrict__ wcomb,
                 const float* __restrict__ b_fc2, float* __restrict__ out)
{
    const int n = blockIdx.x;
    __shared__ float ts[CC * MM];
    for (int idx = threadIdx.x; idx < CC * MM; idx += 256)
        ts[idx] = t[n * CC * MM + idx];
    __syncthreads();
    for (int idx = threadIdx.x; idx < MM * CC; idx += 256) {
        int m = idx % MM, e = idx / MM;
        int l = (m == 0) ? 0 : (m < 4 ? 1 : 2);
        const float* wrow = wcomb + (l * 64 + e) * 64;
        float s = (m == 0) ? b_fc2[e] : 0.f;
#pragma unroll 8
        for (int c = 0; c < 64; ++c) s += ts[c * MM + m] * wrow[c];
        out[(n * MM + m) * CC + e] = s;
    }
}

// ---------------------------------------------------------------------------
extern "C" void kernel_launch(void* const* d_in, const int* in_sizes, int n_in,
                              void* d_out, int out_size, void* d_ws, size_t ws_size,
                              hipStream_t stream)
{
    const float* irreps_x = (const float*)d_in[0];
    const int*   anum     = (const int*)  d_in[1];
    const float* w_fc1    = (const float*)d_in[2];
    const float* b_fc1    = (const float*)d_in[3];
    const float* U3       = (const float*)d_in[4];
    const float* W3       = (const float*)d_in[5];
    const float* U2       = (const float*)d_in[6];
    const float* W2       = (const float*)d_in[7];
    const float* U1       = (const float*)d_in[8];
    const float* W1       = (const float*)d_in[9];
    const float* w_lin    = (const float*)d_in[10];
    const float* w_fc2    = (const float*)d_in[11];
    const float* b_fc2    = (const float*)d_in[12];
    float* out = (float*)d_out;

    // workspace carve-up (all 256B aligned)
    char* ws = (char*)d_ws;
    short* B     = (short*)(ws);                               // 736*160*2 = 235,520
    float* xw    = (float*)(ws + 235520);                      // 1024*64*9*4 = 2,359,296
    float* t     = (float*)(ws + 235520 + 2359296);            // 2,359,296
    float* wcomb = (float*)(ws + 235520 + 2359296 + 2359296);  // 49,152

    build_B     <<<(NCOLP * NK + 255) / 256, 256, 0, stream>>>(U3, U2, B);
    build_wcomb <<<(3 * 64 * 64 + 255) / 256, 256, 0, stream>>>(w_lin, w_fc2, wcomb);
    fc1_kernel  <<<NN, 256, 0, stream>>>(irreps_x, w_fc1, b_fc1, xw);
    mace_body3_main<<<NN * 4, 128, 0, stream>>>(anum, W3, W2, W1, U1, xw, B, t);
    tail_kernel <<<NN, 256, 0, stream>>>(t, wcomb, b_fc2, out);
}
